// CompLinear2_12257836663257
// MI455X (gfx1250) — matrix operand order repaired and checked by reading, not compiled, not run on hardware
//
#include <hip/hip_runtime.h>
#include <hip/hip_bf16.h>

typedef __bf16 bf16_t;
typedef __attribute__((ext_vector_type(16))) __bf16 v16bf;
typedef __attribute__((ext_vector_type(8)))  __bf16 v8bf;
typedef __attribute__((ext_vector_type(8)))  float   v8f;
typedef __attribute__((ext_vector_type(4))) unsigned int u32x4;
typedef __attribute__((ext_vector_type(8))) int          i32x8;
typedef __attribute__((ext_vector_type(4))) int          i32x4;

#define KC   64         // K elements staged per LDS buffer (2 WMMA sub-steps)
#define LDK  72         // LDS K-stride in bf16 (64 data + 8 pad -> 144 B rows)
#define LDW  132        // padded f32 stride for the 128-wide w buffer
#define LDY  65         // padded f32 stride for the 64-wide y buffer

#if defined(__has_builtin)
#if __has_builtin(__builtin_amdgcn_tensor_load_to_lds)
#define HAVE_TDM 1
#endif
#endif
#ifndef HAVE_TDM
#define HAVE_TDM 0
#endif

// -------------------------------------------------------------------------
// WMMA fragment loaders (bf16 16x16x32) -- layouts per CDNA5 ISA.
//  A 16x32: lanes 0-15 -> row=lane, K {0..7,16..23}; lanes 16-31 -> row=lane-16,
//           K {8..15,24..31}.
//  B 32x16 from [N][K] storage: lane%16 -> column N, same K order as A.
// -------------------------------------------------------------------------
__device__ __forceinline__ v16bf frag_a_rowK(const bf16_t* rowptr, int half) {
  v8bf lo = *(const v8bf*)(rowptr + half * 8);
  v8bf hi = *(const v8bf*)(rowptr + 16 + half * 8);
  v16bf r;
#pragma unroll
  for (int i = 0; i < 8; ++i) { r[i] = lo[i]; r[i + 8] = hi[i]; }
  return r;
}

__device__ __forceinline__ v16bf frag_b_colK(const bf16_t* colptr, int half) {
  v8bf lo = *(const v8bf*)(colptr + half * 8);
  v8bf hi = *(const v8bf*)(colptr + 16 + half * 8);
  v16bf r;
#pragma unroll
  for (int i = 0; i < 8; ++i) { r[i] = lo[i]; r[i + 8] = hi[i]; }
  return r;
}

__device__ __forceinline__ uint32_t lds_offset(const void* p) {
  return (uint32_t)(uintptr_t)p;   // low 32 bits of a generic LDS pointer = LDS offset
}

// -------------------------------------------------------------------------
// TDM tile feed: ROWS x 64 bf16 tile, row stride src_ld elements, into LDS with
// a hardware-inserted 16B pad every 128B row  ==> LDS stride LDK (72 bf16).
// D# packing per CDNA5 ISA.
// builtin arity (clang-23): (g0 u32x4, g1 i32x8, g2 i32x4, g3 i32x4, i32x8, cpol).
// -------------------------------------------------------------------------
#if HAVE_TDM
template <int ROWS>
__device__ __forceinline__ void tdm_load_tile(uint32_t lds_off, const bf16_t* gsrc,
                                              uint32_t src_ld) {
  uint64_t ga = (uint64_t)(uintptr_t)gsrc;
  u32x4 g0;
  g0[0] = 1u;                                   // count=1, user descriptor
  g0[1] = lds_off;                              // lds_addr
  g0[2] = (uint32_t)ga;                         // global_addr[31:0]
  g0[3] = (uint32_t)(ga >> 32) | (2u << 30);    // global_addr[56:32] | type=2
  i32x8 g1;
  const uint32_t dim0 = KC, dim1 = ROWS, tile0 = KC, tile1 = ROWS;
  g1[0] = (int)((1u << 16)        // data_size = 2 bytes
              | (1u << 20)        // pad_enable
              | (4u << 22)        // pad_interval: every 32 DWORDs (128 B)
              | (3u << 25));      // pad_amount:   4 DWORDs (16 B)
  g1[1] = (int)((dim0 & 0xffffu) << 16);                   // tensor_dim0[15:0]
  g1[2] = (int)((dim0 >> 16) | ((dim1 & 0xffffu) << 16));  // dim0[31:16] | dim1[15:0]
  g1[3] = (int)((dim1 >> 16) | (tile0 << 16));             // dim1[31:16] | tile_dim0
  g1[4] = (int)tile1;                                      // tile_dim1 (tile_dim2=0)
  g1[5] = (int)src_ld;                                     // tensor_dim0_stride[31:0]
  g1[6] = 0;
  g1[7] = 0;
  i32x4 gz4 = {0, 0, 0, 0};
  i32x8 gz8 = {0, 0, 0, 0, 0, 0, 0, 0};
  __builtin_amdgcn_tensor_load_to_lds(g0, g1, gz4, gz4, gz8, 0);
}
#endif

// All-thread fallback staging (used when TDM builtin is unavailable).
template <int ROWS>
__device__ __forceinline__ void stage_tile(bf16_t* dst, const bf16_t* src,
                                           size_t src_ld, int tid) {
  int r  = tid >> 3;        // 0..31
  int ko = (tid & 7) * 8;   // 16-byte chunks across 128 B row
#pragma unroll
  for (int rep = 0; rep < ROWS / 32; ++rep) {
    *(v8bf*)(dst + (size_t)(r + rep * 32) * LDK + ko) =
        *(const v8bf*)(src + (size_t)(r + rep * 32) * src_ld + ko);
  }
}

// -------------------------------------------------------------------------
// Fused LDLQ step. 64-row tiles, grid = m/64, 256 threads (8 waves).
// Phase 1: acc = E[:, e:e+K] @ L[e:e+K, s:e]   (bf16 WMMA, f32 acc, TDM feeds)
// Phase 2-4: f32 codec (y = (w/rn)@We, y_hat = rint(y), x_hat = y_hat@Wd)
// Emits Wb = bf16(x_hat*rn) and Eb = bf16(weight - x_hat).
// -------------------------------------------------------------------------
__global__ void ldlq_step_kernel(const float* __restrict__ weightF,
                                 const float* __restrict__ row_norm,
                                 const float* __restrict__ We,
                                 const float* __restrict__ Wd,
                                 const bf16_t* __restrict__ Ltb,  // Lt[c][k]=L[k][c]
                                 bf16_t* __restrict__ Eb,
                                 bf16_t* __restrict__ Wb,
                                 int n, int s, int K) {
  extern __shared__ char smem[];
  const int e     = s + 128;
  const int tid   = threadIdx.x;
  const int lane  = tid & 31;          // wave32
  const int wave  = tid >> 5;
  const int half  = lane >> 4;
  const int lr    = lane & 15;
  const int waveM = wave & 3;          // 4 row groups of 16
  const int waveN = wave >> 2;         // 2 col groups of 64
  const size_t rowBase = (size_t)blockIdx.x * 64;

  bf16_t* tA = (bf16_t*)smem;                         // 2 x  64 x LDK
  bf16_t* tB = (bf16_t*)smem + 2 * 64 * LDK;          // 2 x 128 x LDK

  v8f acc[4];
  const v8f vzero = {0.f, 0.f, 0.f, 0.f, 0.f, 0.f, 0.f, 0.f};
#pragma unroll
  for (int ni = 0; ni < 4; ++ni) acc[ni] = vzero;

  if (K > 0) {
#if HAVE_TDM
    if (wave == 0) {
      tdm_load_tile<64>(lds_offset(tA), Eb + rowBase * n + e, (uint32_t)n);
      tdm_load_tile<128>(lds_offset(tB), Ltb + (size_t)s * n + e, (uint32_t)n);
    }
    __builtin_amdgcn_s_wait_tensorcnt(0);   // no-op for waves with TENSORcnt==0
#else
    stage_tile<64>(tA, Eb + rowBase * n + e, n, tid);
    stage_tile<128>(tB, Ltb + (size_t)s * n + e, n, tid);
#endif
    __syncthreads();
    int buf = 0;
    for (int kk = 0; kk < K; kk += KC) {
      int nbuf = buf ^ 1;
      if (kk + KC < K) {
#if HAVE_TDM
        if (wave == 0) {
          tdm_load_tile<64>(lds_offset(tA) + nbuf * 64 * LDK * 2,
                            Eb + rowBase * n + e + kk + KC, (uint32_t)n);
          tdm_load_tile<128>(lds_offset(tB) + nbuf * 128 * LDK * 2,
                             Ltb + (size_t)s * n + e + kk + KC, (uint32_t)n);
        }
#else
        stage_tile<64>(tA + nbuf * 64 * LDK, Eb + rowBase * n + e + kk + KC, n, tid);
        stage_tile<128>(tB + nbuf * 128 * LDK, Ltb + (size_t)s * n + e + kk + KC, n, tid);
#endif
      }
      const bf16_t* A = tA + buf * 64 * LDK;
      const bf16_t* B = tB + buf * 128 * LDK;
#pragma unroll
      for (int sub = 0; sub < 2; ++sub) {       // two 32-K sub-steps per buffer
        int kb = sub * 32;
        v16bf af = frag_a_rowK(A + (size_t)(waveM * 16 + lr) * LDK + kb, half);
        v16bf bfr[4];
#pragma unroll
        for (int ni = 0; ni < 4; ++ni)
          bfr[ni] = frag_b_colK(B + (size_t)(waveN * 64 + ni * 16 + lr) * LDK + kb, half);
#pragma unroll
        for (int ni = 0; ni < 4; ++ni)
          acc[ni] = __builtin_amdgcn_wmma_f32_16x16x32_bf16(
              false, af, false, bfr[ni], (short)0, acc[ni], false, false);
      }
#if HAVE_TDM
      __builtin_amdgcn_s_wait_tensorcnt(0);
#endif
      __syncthreads();
      buf = nbuf;
    }
  }

  // ---- phase 2: w = weight[:, s:e] + acc -> LDS f32 (full precision)
  float* wbuf = (float*)smem;                                   // 64 x LDW
  float* ybuf = (float*)(smem + 64 * LDW * sizeof(float));      // 64 x LDY
  __syncthreads();
#pragma unroll
  for (int ni = 0; ni < 4; ++ni) {
    int col = waveN * 64 + ni * 16 + lr;
#pragma unroll
    for (int i = 0; i < 8; ++i) {
      int r = waveM * 16 + half * 8 + i;    // C layout: VGPR i -> M=i / M=8+i
      wbuf[r * LDW + col] = acc[ni][i] + weightF[(rowBase + r) * (size_t)n + s + col];
    }
  }
  __syncthreads();

  // ---- phase 3: y_hat = rint((w / row_norm) @ We)   (RNE == jnp.round)
  {
    int l  = tid & 63;
    int rg = tid >> 6;            // 0..3 -> 16 rows each
    for (int rr = 0; rr < 16; ++rr) {
      int r = rg * 16 + rr;
      float a = 0.f;
#pragma unroll 8
      for (int k = 0; k < 128; ++k) a = fmaf(wbuf[r * LDW + k], We[k * 64 + l], a);
      float y = a / row_norm[rowBase + r];
      ybuf[r * LDY + l] = rintf(y);
    }
  }
  __syncthreads();

  // ---- phase 4: x_hat = y_hat @ Wd; emit Wb (scaled) and Eb (error)
  {
    int c  = tid & 127;
    int cg = tid >> 7;            // 0..1 -> 32 rows each
    for (int rr = 0; rr < 32; ++rr) {
      int r = cg * 32 + rr;
      float a = 0.f;
#pragma unroll 8
      for (int l = 0; l < 64; ++l) a = fmaf(ybuf[r * LDY + l], Wd[l * 128 + c], a);
      size_t idx = (rowBase + r) * (size_t)n + s + c;
      float rn = row_norm[rowBase + r];
      Wb[idx] = (bf16_t)(a * rn);
      Eb[idx] = (bf16_t)(weightF[idx] - a);
    }
  }
}

// -------------------------------------------------------------------------
// out = x @ (W_hat*row_norm)^T + bias. 128x128 tile/WG, 8 waves x (32x64),
// double-buffered LDS (K=64 per buffer) with TDM tile feeds.
// -------------------------------------------------------------------------
__global__ void out_gemm_kernel(const bf16_t* __restrict__ Xb,
                                const bf16_t* __restrict__ Wb,
                                const float* __restrict__ bias,
                                float* __restrict__ out,
                                int n /*K*/, int m /*out cols*/) {
  extern __shared__ char smem[];
  const int tid   = threadIdx.x;
  const int lane  = tid & 31;
  const int wave  = tid >> 5;
  const int half  = lane >> 4;
  const int lr    = lane & 15;
  const int waveM = wave & 3;
  const int waveN = wave >> 2;
  const size_t rowBase = (size_t)blockIdx.y * 128;   // batch rows
  const size_t colBase = (size_t)blockIdx.x * 128;   // output cols (rows of Wb)

  bf16_t* tA = (bf16_t*)smem;                        // 2 x 128 x LDK
  bf16_t* tB = (bf16_t*)smem + 2 * 128 * LDK;        // 2 x 128 x LDK

  v8f acc[2][4];
  const v8f vzero = {0.f, 0.f, 0.f, 0.f, 0.f, 0.f, 0.f, 0.f};
#pragma unroll
  for (int mi = 0; mi < 2; ++mi)
#pragma unroll
    for (int ni = 0; ni < 4; ++ni) acc[mi][ni] = vzero;

#if HAVE_TDM
  if (wave == 0) {
    tdm_load_tile<128>(lds_offset(tA), Xb + rowBase * n, (uint32_t)n);
    tdm_load_tile<128>(lds_offset(tB), Wb + colBase * n, (uint32_t)n);
  }
  __builtin_amdgcn_s_wait_tensorcnt(0);
#else
  stage_tile<128>(tA, Xb + rowBase * n, n, tid);
  stage_tile<128>(tB, Wb + colBase * n, n, tid);
#endif
  __syncthreads();
  int buf = 0;
  for (int kk = 0; kk < n; kk += KC) {
    int nbuf = buf ^ 1;
    if (kk + KC < n) {
#if HAVE_TDM
      if (wave == 0) {
        tdm_load_tile<128>(lds_offset(tA) + nbuf * 128 * LDK * 2,
                           Xb + rowBase * n + kk + KC, (uint32_t)n);
        tdm_load_tile<128>(lds_offset(tB) + nbuf * 128 * LDK * 2,
                           Wb + colBase * n + kk + KC, (uint32_t)n);
      }
#else
      stage_tile<128>(tA + nbuf * 128 * LDK, Xb + rowBase * n + kk + KC, n, tid);
      stage_tile<128>(tB + nbuf * 128 * LDK, Wb + colBase * n + kk + KC, n, tid);
#endif
    }
    const bf16_t* A = tA + buf * 128 * LDK;
    const bf16_t* B = tB + buf * 128 * LDK;
#pragma unroll
    for (int sub = 0; sub < 2; ++sub) {
      int kb = sub * 32;
      v16bf af[2], bfr[4];
#pragma unroll
      for (int mi = 0; mi < 2; ++mi)
        af[mi] = frag_a_rowK(A + (size_t)(waveM * 32 + mi * 16 + lr) * LDK + kb, half);
#pragma unroll
      for (int ni = 0; ni < 4; ++ni)
        bfr[ni] = frag_b_colK(B + (size_t)(waveN * 64 + ni * 16 + lr) * LDK + kb, half);
#pragma unroll
      for (int mi = 0; mi < 2; ++mi)
#pragma unroll
        for (int ni = 0; ni < 4; ++ni)
          acc[mi][ni] = __builtin_amdgcn_wmma_f32_16x16x32_bf16(
              false, af[mi], false, bfr[ni], (short)0, acc[mi][ni], false, false);
    }
#if HAVE_TDM
    __builtin_amdgcn_s_wait_tensorcnt(0);
#endif
    __syncthreads();
    buf = nbuf;
  }

#pragma unroll
  for (int mi = 0; mi < 2; ++mi) {
#pragma unroll
    for (int ni = 0; ni < 4; ++ni) {
      int c = waveN * 64 + ni * 16 + lr;
#pragma unroll
      for (int i = 0; i < 8; ++i) {
        int r = waveM * 32 + mi * 16 + half * 8 + i;
        out[(rowBase + r) * (size_t)m + colBase + c] =
            acc[mi][ni][i] + bias[colBase + c];
      }
    }
  }
}

// -------------------------------------------------------------------------
// Helpers: f32 -> bf16 convert; f32 -> bf16 transposed copy (for L).
// -------------------------------------------------------------------------
__global__ void f32_to_bf16_kernel(const float* __restrict__ in,
                                   bf16_t* __restrict__ out, size_t count) {
  size_t i = (size_t)blockIdx.x * blockDim.x + threadIdx.x;
  size_t stride = (size_t)gridDim.x * blockDim.x;
  for (; i < count; i += stride) out[i] = (bf16_t)in[i];
}

__global__ void transpose_to_bf16_kernel(const float* __restrict__ L,
                                         bf16_t* __restrict__ Lt, int n) {
  __shared__ float tile[32][33];
  int x = blockIdx.x * 32 + threadIdx.x;   // col of L
  int y = blockIdx.y * 32 + threadIdx.y;   // row of L
#pragma unroll
  for (int j = 0; j < 32; j += 8)
    tile[threadIdx.y + j][threadIdx.x] = L[(size_t)(y + j) * n + x];
  __syncthreads();
  int tx = blockIdx.y * 32 + threadIdx.x;  // Lt col = L row
  int ty = blockIdx.x * 32 + threadIdx.y;  // Lt row = L col
#pragma unroll
  for (int j = 0; j < 32; j += 8)
    Lt[(size_t)(ty + j) * n + tx] = (bf16_t)tile[threadIdx.x][threadIdx.y + j];
}

// -------------------------------------------------------------------------
extern "C" void kernel_launch(void* const* d_in, const int* in_sizes, int n_in,
                              void* d_out, int out_size, void* d_ws, size_t ws_size,
                              hipStream_t stream) {
  const float* x        = (const float*)d_in[0];
  const float* weight   = (const float*)d_in[1];
  const float* bias     = (const float*)d_in[2];
  const float* row_norm = (const float*)d_in[3];
  const float* L        = (const float*)d_in[4];
  const float* We       = (const float*)d_in[5];
  const float* Wd       = (const float*)d_in[6];
  float* out = (float*)d_out;

  const int m  = in_sizes[2];            // 4096
  const int n  = in_sizes[1] / m;        // 4096
  const int Bn = in_sizes[0] / n;        // 4096
  const int BS = 128;
  const int nb = n / BS;

  // workspace layout (bf16): Lt | E | W_scaled | Xb
  bf16_t* Ltb = (bf16_t*)d_ws;
  bf16_t* Eb  = Ltb + (size_t)n * n;
  bf16_t* Wb  = Eb + (size_t)m * n;
  bf16_t* Xb  = Wb + (size_t)m * n;

  f32_to_bf16_kernel<<<2048, 256, 0, stream>>>(x, Xb, (size_t)Bn * n);
  {
    dim3 b(32, 8), g(n / 32, n / 32);
    transpose_to_bf16_kernel<<<g, b, 0, stream>>>(L, Ltb, n);
  }

  // sequential LDLQ chain, last block first (matches the reference scan)
  size_t smem_gemm_step = (size_t)2 * (64 + 128) * LDK * sizeof(bf16_t);
  size_t smem_codec     = (size_t)64 * LDW * sizeof(float)
                        + (size_t)64 * LDY * sizeof(float);
  const size_t SMEM_STEP = smem_gemm_step > smem_codec ? smem_gemm_step : smem_codec;
  for (int i = 0; i < nb; ++i) {
    int e = n - i * BS;
    int s = e - BS;
    int K = i * BS;
    ldlq_step_kernel<<<m / 64, 256, SMEM_STEP, stream>>>(
        weight, row_norm, We, Wd, Ltb, Eb, Wb, n, s, K);
  }

  // final out = x @ W^T + bias
  const size_t SMEM_GEMM = (size_t)2 * (128 + 128) * LDK * sizeof(bf16_t);
  dim3 g(m / 128, Bn / 128);
  out_gemm_kernel<<<g, 256, SMEM_GEMM, stream>>>(Xb, Wb, bias, out, n, m);
}
